// CausalSelfAttention_9835475108116
// MI455X (gfx1250) — compile-verified, matrix operand order checked by reading, not run on hardware
//
#include <hip/hip_runtime.h>
#include <hip/hip_bf16.h>

typedef __attribute__((ext_vector_type(16))) __bf16         v16bf;
typedef __attribute__((ext_vector_type(8)))  float          v8f;
typedef __attribute__((ext_vector_type(8)))  unsigned short us8;

#define B_ 4
#define T_ 2048
#define C_ 1024
#define H_ 16
#define D_ 64

struct U4x2 { uint4 a, b; };

static __device__ __forceinline__ unsigned short f2bf(float f) {
  unsigned u = __float_as_uint(f);
  u += 0x7FFFu + ((u >> 16) & 1u);            // round-to-nearest-even
  return (unsigned short)(u >> 16);
}

// Build a 16x? bf16 fragment from two contiguous 16-byte runs at p and p+16
// halves (ISA A/B layout: elements 0..7 = K kbase..+7, 8..15 = kbase+16..+23).
static __device__ __forceinline__ v16bf frag_ld(const unsigned short* p) {
  U4x2 u;
  u.a = *(const uint4*)(p);
  u.b = *(const uint4*)(p + 16);
  return __builtin_bit_cast(v16bf, u);
}

// ---------------------------------------------------------------- cast pass
__global__ void cast_f32_bf16(const float* __restrict__ in,
                              unsigned short* __restrict__ out, int n) {
  int i = blockIdx.x * blockDim.x + threadIdx.x;
  int stride = gridDim.x * blockDim.x;
  for (; i < n; i += stride) out[i] = f2bf(in[i]);
}

// ------------------------------------------------- bf16 WMMA GEMM (64x64 tile)
// mode 0: QKV projection -> scatter bf16 into Q/K/V [B,H,T,D]
// mode 1: final projection -> fp32 Co[M,N] (+bias)
#define BSTRIDE 40   // padded row stride (halves) of transposed B tile; 80B, 16B-aligned
__global__ void __launch_bounds__(128) gemm_bf16_wmma(
    const unsigned short* __restrict__ A,   // [M,K] bf16
    const unsigned short* __restrict__ Bm,  // [K,N] bf16
    const float* __restrict__ bias,         // [N]
    int M, int K, int N, int mode,
    unsigned short* __restrict__ Qo, unsigned short* __restrict__ Ko,
    unsigned short* __restrict__ Vo, float* __restrict__ Co)
{
  __shared__ __align__(16) unsigned short Al[64 * 32];        // [m][k]
  __shared__ __align__(16) unsigned short Blt[64 * BSTRIDE];  // [n][k] (transposed)

  const int tid  = threadIdx.x;
  const int wave = tid >> 5;
  const int lane = tid & 31;
  const int mb   = blockIdx.y * 64;
  const int nb   = blockIdx.x * 64;
  const int lrow = lane & 15;
  const int hi   = lane >> 4;
  const int kbase = hi * 8;                 // ISA 16-bit A/B half-wave K split

  v8f acc[4];
  #pragma unroll
  for (int i = 0; i < 4; ++i) acc[i] = v8f{};

  // staging thread mapping
  const int arow = tid >> 1, acolb = (tid & 1) * 16;   // A: 64x32, 16 halves/thread
  const int kpair = tid >> 3, ncg = tid & 7;           // B: 16 k-pairs x 8 col-groups

  for (int kk = 0; kk < K; kk += 32) {
    {  // stage A block 64x32 (row-major, contiguous)
      const uint4* src = (const uint4*)(A + (size_t)(mb + arow) * K + kk + acolb);
      uint4* dst = (uint4*)(Al + arow * 32 + acolb);
      dst[0] = src[0]; dst[1] = src[1];
    }
    {  // stage B block 32x64, transposed into Blt[n][k] with packed b32 stores
      const int k0 = kk + kpair * 2;
      const int n0 = ncg * 8;
      us8 r0 = *(const us8*)(Bm + (size_t)k0 * N + nb + n0);
      us8 r1 = *(const us8*)(Bm + (size_t)(k0 + 1) * N + nb + n0);
      #pragma unroll
      for (int j = 0; j < 8; ++j) {
        unsigned pack = (unsigned)r0[j] | ((unsigned)r1[j] << 16);
        *(unsigned*)(Blt + (n0 + j) * BSTRIDE + kpair * 2) = pack;
      }
    }
    __syncthreads();

    v16bf afrag = frag_ld(Al + (wave * 16 + lrow) * 32 + kbase);
    #pragma unroll
    for (int nt = 0; nt < 4; ++nt) {
      v16bf bfrag = frag_ld(Blt + (nt * 16 + lrow) * BSTRIDE + kbase);
      acc[nt] = __builtin_amdgcn_wmma_f32_16x16x32_bf16(
          false, afrag, false, bfrag, (short)0, acc[nt], false, false);
    }
    __syncthreads();
  }

  #pragma unroll
  for (int nt = 0; nt < 4; ++nt) {
    #pragma unroll
    for (int r = 0; r < 8; ++r) {
      int m = mb + wave * 16 + r + 8 * hi;
      int n = nb + nt * 16 + lrow;
      float v = acc[nt][r] + bias[n];
      if (mode == 0) {
        int b = m / T_, t = m % T_;
        int which = n >> 10, c = n & 1023, h = c >> 6, d = c & 63;
        size_t idx = ((size_t)(b * H_ + h) * T_ + t) * D_ + d;
        unsigned short bv = f2bf(v);
        if (which == 0)      Qo[idx] = bv;
        else if (which == 1) Ko[idx] = bv;
        else                 Vo[idx] = bv;
      } else {
        Co[(size_t)m * N + n] = v;
      }
    }
  }
}

// -------------------------------------------- flash attention (online softmax)
// grid: (T/64, B*H); block: 128 threads = 4 waves, wave w owns q rows qb0+16w..
__global__ void __launch_bounds__(128) flash_attn_wmma(
    const unsigned short* __restrict__ Qb,
    const unsigned short* __restrict__ Kb,
    const unsigned short* __restrict__ Vb,
    unsigned short* __restrict__ Y)     // [B,T,C] bf16
{
  __shared__ __align__(16) unsigned short Kl[32 * 64];        // [kv][d]
  __shared__ __align__(16) unsigned short Vt[64 * BSTRIDE];   // [d][kv] (transposed)
  __shared__ __align__(16) unsigned short Pl[4][16 * 32];     // per-wave P [m][kv]

  const int tid  = threadIdx.x;
  const int wave = tid >> 5;
  const int lane = tid & 31;
  const int bh   = blockIdx.y;           // b*H + h
  const int qb0  = blockIdx.x * 64;
  const int qb   = qb0 + wave * 16;
  const int lrow = lane & 15;
  const int hi   = lane >> 4;
  const int kbase = hi * 8;
  const size_t base = (size_t)bh * T_ * D_;

  // Q fragments: two K=32 chunks covering D=64 (global b128 loads)
  const unsigned short* qrow = Qb + base + (size_t)(qb + lrow) * D_;
  v16bf qfrag[2];
  qfrag[0] = frag_ld(qrow + 0 * 32 + kbase);
  qfrag[1] = frag_ld(qrow + 1 * 32 + kbase);

  v8f o[4];
  #pragma unroll
  for (int i = 0; i < 4; ++i) o[i] = v8f{};
  float mrow[8], lsum[8];
  #pragma unroll
  for (int r = 0; r < 8; ++r) { mrow[r] = -3.0e38f; lsum[r] = 0.0f; }

  const int nchunk = (qb0 + 64) / 32;    // uniform per block
  const float sscale = 0.125f;           // 1/sqrt(D)

  // staging thread mappings
  const int krow = tid >> 2, kcolb = (tid & 3) * 16;   // K: 32x64 row-major
  const int kvpair = tid >> 3, dcg = tid & 7;          // V: 16 kv-pairs x 8 d-groups

  for (int j = 0; j < nchunk; ++j) {
    const int kv0 = j * 32;
    {  // K chunk [32][64] row-major (d innermost; matches S B-frag layout)
      const uint4* ks = (const uint4*)(Kb + base + (size_t)(kv0 + krow) * D_ + kcolb);
      uint4* kd = (uint4*)(Kl + krow * 64 + kcolb);
      kd[0] = ks[0]; kd[1] = ks[1];
    }
    {  // V chunk transposed into Vt[d][kv] with packed b32 stores
      const int kvl = kvpair * 2;
      const int d0  = dcg * 8;
      us8 r0 = *(const us8*)(Vb + base + (size_t)(kv0 + kvl) * D_ + d0);
      us8 r1 = *(const us8*)(Vb + base + (size_t)(kv0 + kvl + 1) * D_ + d0);
      #pragma unroll
      for (int jj = 0; jj < 8; ++jj) {
        unsigned pack = (unsigned)r0[jj] | ((unsigned)r1[jj] << 16);
        *(unsigned*)(Vt + (d0 + jj) * BSTRIDE + kvl) = pack;
      }
    }
    __syncthreads();

    if (kv0 <= qb + 15) {                // chunk intersects this wave's rows
      // S = Q K^T : two 16x16 S tiles (kv 0..15, 16..31), k over D
      v8f s0 = v8f{}, s1 = v8f{};
      #pragma unroll
      for (int c = 0; c < 2; ++c) {
        v16bf bf0 = frag_ld(Kl + lrow        * 64 + c * 32 + kbase);
        v16bf bf1 = frag_ld(Kl + (16 + lrow) * 64 + c * 32 + kbase);
        s0 = __builtin_amdgcn_wmma_f32_16x16x32_bf16(
            false, qfrag[c], false, bf0, (short)0, s0, false, false);
        s1 = __builtin_amdgcn_wmma_f32_16x16x32_bf16(
            false, qfrag[c], false, bf1, (short)0, s1, false, false);
      }

      // scale + causal mask (C-layout: row = r + 8*hi, col = lrow)
      float s0f[8], s1f[8];
      #pragma unroll
      for (int r = 0; r < 8; ++r) {
        int q = qb + r + 8 * hi;
        s0f[r] = (kv0 + lrow      <= q) ? s0[r] * sscale : -3.0e38f;
        s1f[r] = (kv0 + 16 + lrow <= q) ? s1[r] * sscale : -3.0e38f;
      }

      // online softmax row stats (rows live across a 16-lane half-wave)
      float mnew[8], scl[8];
      #pragma unroll
      for (int r = 0; r < 8; ++r) {
        float v = fmaxf(s0f[r], s1f[r]);
        #pragma unroll
        for (int msk = 1; msk < 16; msk <<= 1) v = fmaxf(v, __shfl_xor(v, msk, 16));
        mnew[r] = fmaxf(mrow[r], v);
        scl[r]  = __expf(mrow[r] - mnew[r]);
        mrow[r] = mnew[r];
      }
      float p0[8], p1[8];
      #pragma unroll
      for (int r = 0; r < 8; ++r) {
        p0[r] = __expf(s0f[r] - mnew[r]);
        p1[r] = __expf(s1f[r] - mnew[r]);
        float s = p0[r] + p1[r];
        #pragma unroll
        for (int msk = 1; msk < 16; msk <<= 1) s += __shfl_xor(s, msk, 16);
        lsum[r] = lsum[r] * scl[r] + s;
      }
      #pragma unroll
      for (int nt = 0; nt < 4; ++nt) {
        #pragma unroll
        for (int r = 0; r < 8; ++r) o[nt][r] *= scl[r];
      }

      // C-layout -> A-layout transpose of P through per-wave LDS
      unsigned short* pw = Pl[wave];
      #pragma unroll
      for (int r = 0; r < 8; ++r) {
        int m = r + 8 * hi;
        pw[m * 32 + lrow]      = f2bf(p0[r]);
        pw[m * 32 + 16 + lrow] = f2bf(p1[r]);
      }
      asm volatile("s_wait_dscnt 0" ::: "memory");
      v16bf pfrag = frag_ld(pw + lrow * 32 + kbase);

      // O += P @ V  (k = 32 kv positions, 4 N-tiles over D)
      #pragma unroll
      for (int nt = 0; nt < 4; ++nt) {
        v16bf vfrag = frag_ld(Vt + (nt * 16 + lrow) * BSTRIDE + kbase);
        o[nt] = __builtin_amdgcn_wmma_f32_16x16x32_bf16(
            false, pfrag, false, vfrag, (short)0, o[nt], false, false);
      }
    }
    __syncthreads();
  }

  // normalize and write Y [B,T,C] bf16
  const int b = bh / H_, h = bh % H_;
  #pragma unroll
  for (int nt = 0; nt < 4; ++nt) {
    #pragma unroll
    for (int r = 0; r < 8; ++r) {
      int t = qb + r + 8 * hi;
      float v = o[nt][r] / lsum[r];
      Y[(size_t)(b * T_ + t) * C_ + h * D_ + nt * 16 + lrow] = f2bf(v);
    }
  }
}

// ------------------------------------------------------------------ launcher
extern "C" void kernel_launch(void* const* d_in, const int* in_sizes, int n_in,
                              void* d_out, int out_size, void* d_ws, size_t ws_size,
                              hipStream_t stream) {
  const float* x      = (const float*)d_in[0];
  const float* w_qkv  = (const float*)d_in[1];
  const float* b_qkv  = (const float*)d_in[2];
  const float* w_proj = (const float*)d_in[3];
  const float* b_proj = (const float*)d_in[4];
  float* out = (float*)d_out;

  char* ws = (char*)d_ws;
  size_t off = 0;
  auto alloc_us = [&](size_t elems) {
    unsigned short* p = (unsigned short*)(ws + off);
    off += (elems * sizeof(unsigned short) + 255) & ~(size_t)255;
    return p;
  };

  const size_t nx  = (size_t)B_ * T_ * C_;        // 8388608
  const size_t nwq = (size_t)C_ * 3 * C_;         // 3145728
  const size_t nwp = (size_t)C_ * C_;             // 1048576
  const size_t nqkv = (size_t)B_ * H_ * T_ * D_;  // 8388608

  unsigned short* xb  = alloc_us(nx);
  unsigned short* wqb = alloc_us(nwq);
  unsigned short* wpb = alloc_us(nwp);
  unsigned short* Qb  = alloc_us(nqkv);
  unsigned short* Kb  = alloc_us(nqkv);
  unsigned short* Vb  = alloc_us(nqkv);
  unsigned short* Yb  = alloc_us(nx);

  // 1) casts to bf16
  cast_f32_bf16<<<2048, 256, 0, stream>>>(x,      xb,  (int)nx);
  cast_f32_bf16<<<2048, 256, 0, stream>>>(w_qkv,  wqb, (int)nwq);
  cast_f32_bf16<<<1024, 256, 0, stream>>>(w_proj, wpb, (int)nwp);

  // 2) QKV projection GEMM: [8192,1024] x [1024,3072] -> Q/K/V [B,H,T,D]
  {
    dim3 grid(3 * C_ / 64, (B_ * T_) / 64);
    gemm_bf16_wmma<<<grid, 128, 0, stream>>>(
        xb, wqb, b_qkv, B_ * T_, C_, 3 * C_, /*mode=*/0, Qb, Kb, Vb, nullptr);
  }

  // 3) flash attention -> Y [B,T,C] bf16
  {
    dim3 grid(T_ / 64, B_ * H_);
    flash_attn_wmma<<<grid, 128, 0, stream>>>(Qb, Kb, Vb, Yb);
  }

  // 4) output projection GEMM: [8192,1024] x [1024,1024] -> fp32 out
  {
    dim3 grid(C_ / 64, (B_ * T_) / 64);
    gemm_bf16_wmma<<<grid, 128, 0, stream>>>(
        Yb, wpb, b_proj, B_ * T_, C_, C_, /*mode=*/1,
        nullptr, nullptr, nullptr, out);
  }
}
